// SUE_33328946217337
// MI455X (gfx1250) — compile-verified, matrix operand order checked by reading, not run on hardware
//
#include <hip/hip_runtime.h>
#include <hip/hip_bf16.h>

typedef __bf16 bf16;
typedef __attribute__((ext_vector_type(16))) __bf16 v16bf;
typedef __attribute__((ext_vector_type(8)))  __bf16 v8bf;
typedef __attribute__((ext_vector_type(8)))  float  v8f;

// ---- problem constants (match reference setup_inputs) ----
constexpr int B_   = 64;
constexpr int NEWS = 50;
constexpr int H_   = 100;
constexpr int C_   = 19;    // C0 + 1
constexpr int N_   = 118;   // H + C0
constexpr int D_   = 400;
constexpr int A_   = 200;
constexpr int L_   = 2;

// padded dims: M/N tiles of 16, K tiles of 32
constexpr int NP_G  = 128;  // pad(118) for graph M and K
constexpr int DP    = 416;  // pad(400) as K
constexpr int AP_N  = 208;  // pad(200) as N
constexpr int AP_K  = 224;  // pad(200) as K
constexpr int HP    = 112;  // pad(100) as N
constexpr int NEWSP = 64;   // pad(50)  as M

// =====================================================================
// WMMA GEMM on pre-packed bf16 operands.
//   A : [Mp, Kp] bf16 row-major (zero padded)
//   Bt: [Np, Kp] bf16 row-major = B^T (zero padded)
//   C : fp32 [M, N] (ldc), epilogue: scale, +bias, relu, +resid
// One wave per 16x16 C tile. Inner loop: 4x global_load_b128, 1x
// v_wmma_f32_16x16x32_bf16 -- zero predication inside the loop.
// Fragment map (16-bit A 16x32): lane<16 row m=lane K-base k0,
// lane>=16 row m=lane-16 K-base k0+8; elems 0..7 = kb.., 8..15 = kb+16..
// B mirrors with column n = lane&15. C: vgpr r -> (M=r+8*half, N=lane&15).
// =====================================================================
__global__ void __launch_bounds__(32)
wmma_gemm_kernel(const bf16* __restrict__ A, long long sAb,
                 const bf16* __restrict__ Bt, long long sBb, int Kp,
                 const float* __restrict__ bias,
                 const float* __restrict__ resid, long long sRb,
                 float* __restrict__ Cc, long long sCb, int ldc,
                 int M, int Nn, float scale, int do_relu)
{
    const int tm = blockIdx.x * 16;
    const int tn = blockIdx.y * 16;
    const int b  = blockIdx.z;
    A  += (long long)b * sAb;
    Bt += (long long)b * sBb;
    Cc += (long long)b * sCb;
    const float* rp = resid ? resid + (long long)b * sRb : nullptr;

    const int lane = threadIdx.x;
    const int half = lane >> 4;
    const int l16  = lane & 15;

    const bf16* ap = A  + (long long)(tm + l16) * Kp + half * 8;
    const bf16* bp = Bt + (long long)(tn + l16) * Kp + half * 8;

    v8f acc = {};
    for (int k0 = 0; k0 < Kp; k0 += 32) {
        v8bf alo = *(const v8bf*)(ap + k0);
        v8bf ahi = *(const v8bf*)(ap + k0 + 16);
        v8bf blo = *(const v8bf*)(bp + k0);
        v8bf bhi = *(const v8bf*)(bp + k0 + 16);
        v16bf af, bv;
#pragma unroll
        for (int j = 0; j < 8; ++j) {
            af[j] = alo[j]; af[8 + j] = ahi[j];
            bv[j] = blo[j]; bv[8 + j] = bhi[j];
        }
        acc = __builtin_amdgcn_wmma_f32_16x16x32_bf16(
                  false, af, false, bv, (short)0, acc, false, false);
    }

#pragma unroll
    for (int r = 0; r < 8; ++r) {
        const int mr = tm + r + half * 8;
        const int nc = tn + l16;
        if (mr < M && nc < Nn) {
            float v = acc[r] * scale;
            if (bias)    v += bias[nc];
            if (do_relu) v  = fmaxf(v, 0.0f);
            if (rp)      v += rp[(long long)mr * ldc + nc];
            Cc[(long long)mr * ldc + nc] = v;
        }
    }
}

// ---- pack f32 [b,R,Cn] -> bf16 [b,Rp,Cp], zero padded ----
__global__ void __launch_bounds__(256)
pack_bf16_kernel(const float* __restrict__ in, long long sIn,
                 bf16* __restrict__ out, long long sOut,
                 int R, int Cn, int Rp, int Cp, long long total)
{
    long long idx = (long long)blockIdx.x * 256 + threadIdx.x;
    if (idx >= total) return;
    int c = (int)(idx % Cp);
    long long t = idx / Cp;
    int r = (int)(t % Rp);
    int b = (int)(t / Rp);
    float v = (r < R && c < Cn) ? in[(long long)b * sIn + (long long)r * Cn + c] : 0.0f;
    out[(long long)b * sOut + (long long)r * Cp + c] = (bf16)v;
}

// ---- pack transposed: f32 [b,R,Cn] -> bf16 [b,Cp,Rp] (out[c,r]=in[r,c]) ----
__global__ void __launch_bounds__(256)
packT_bf16_kernel(const float* __restrict__ in, long long sIn,
                  bf16* __restrict__ out, long long sOut,
                  int R, int Cn, int Cp, int Rp, long long total)
{
    long long idx = (long long)blockIdx.x * 256 + threadIdx.x;
    if (idx >= total) return;
    int rr = (int)(idx % Rp);            // out col = original row
    long long t = idx / Rp;
    int cc = (int)(t % Cp);              // out row = original col
    int b  = (int)(t / Cp);
    float v = (rr < R && cc < Cn) ? in[(long long)b * sIn + (long long)rr * Cn + cc] : 0.0f;
    out[(long long)b * sOut + (long long)cc * Rp + rr] = (bf16)v;
}

// ---- h0 = concat(history, broadcast(proxy)) ; also seed h_cur ----
__global__ void __launch_bounds__(256)
build_h0_kernel(const float* __restrict__ hist, const float* __restrict__ proxy,
                float* __restrict__ h0, float* __restrict__ hcur)
{
    long long idx = (long long)blockIdx.x * blockDim.x + threadIdx.x;
    const long long total = (long long)B_ * N_ * D_;
    if (idx >= total) return;
    int d = (int)(idx % D_);
    long long t = idx / D_;
    int i = (int)(t % N_);
    int b = (int)(t / N_);
    float v = (i < H_) ? hist[((long long)b * H_ + i) * D_ + d]
                       : proxy[(long long)(i - H_) * D_ + d];
    h0[idx]   = v;
    hcur[idx] = v;
}

// ---- row LayerNorm (D=400), gamma/beta per layer ----
__global__ void __launch_bounds__(256)
layernorm_kernel(const float* __restrict__ in, float* __restrict__ out,
                 const float* __restrict__ gamma, const float* __restrict__ beta)
{
    __shared__ float r1[256], r2[256];
    const int row = blockIdx.x;
    const float* ip = in + (long long)row * D_;
    float s1 = 0.0f, s2 = 0.0f;
    for (int d = threadIdx.x; d < D_; d += 256) {
        float v = ip[d]; s1 += v; s2 += v * v;
    }
    r1[threadIdx.x] = s1; r2[threadIdx.x] = s2;
    __syncthreads();
    for (int s = 128; s > 0; s >>= 1) {
        if (threadIdx.x < s) { r1[threadIdx.x] += r1[threadIdx.x + s];
                               r2[threadIdx.x] += r2[threadIdx.x + s]; }
        __syncthreads();
    }
    const float mean = r1[0] * (1.0f / D_);
    const float var  = r2[0] * (1.0f / D_) - mean * mean;
    const float rs   = rsqrtf(var + 1e-5f);
    float* op = out + (long long)row * D_;
    for (int d = threadIdx.x; d < D_; d += 256)
        op[d] = (ip[d] - mean) * rs * gamma[d] + beta[d];
}

// ---- g = (h + h0)[:, :H, :] ----
__global__ void __launch_bounds__(256)
make_g_kernel(const float* __restrict__ hcur, const float* __restrict__ h0,
              float* __restrict__ g)
{
    long long idx = (long long)blockIdx.x * blockDim.x + threadIdx.x;
    const long long total = (long long)B_ * H_ * D_;
    if (idx >= total) return;
    int d = (int)(idx % D_);
    long long t = idx / D_;
    int h = (int)(t % H_);
    int b = (int)(t / H_);
    long long src = ((long long)b * N_ + h) * D_ + d;
    g[idx] = hcur[src] + h0[src];
}

// ---- segment (per-category) softmax over H for each (b, news) ----
__global__ void __launch_bounds__(128)
segment_softmax_kernel(const float* __restrict__ a, const int* __restrict__ cidx,
                       float* __restrict__ alpha)
{
    __shared__ float sa[H_];
    __shared__ int   sci[H_];
    __shared__ float cmax[C_];
    __shared__ float csum[C_];
    const int bn = blockIdx.x;
    const int b  = bn / NEWS;
    const int tid = threadIdx.x;
    for (int i = tid; i < H_; i += 128) {
        sa[i]  = a[(long long)bn * H_ + i];
        sci[i] = cidx[(long long)b * H_ + i];
    }
    __syncthreads();
    if (tid < C_) {
        float mx = -1e30f;
        for (int i = 0; i < H_; ++i) if (sci[i] == tid) mx = fmaxf(mx, sa[i]);
        cmax[tid] = mx;
    }
    __syncthreads();
    if (tid < C_) {
        float s = 0.0f;
        for (int i = 0; i < H_; ++i) if (sci[i] == tid) s += __expf(sa[i] - cmax[tid]);
        csum[tid] = s;
    }
    __syncthreads();
    for (int i = tid; i < H_; i += 128) {
        int c = sci[i];
        alpha[(long long)bn * H_ + i] = __expf(sa[i] - cmax[c]) / csum[c];
    }
}

// ---- scatter-sum: intra[b,n,c,:] = sum_{h: cat=c} alpha[b,n,h] * g[b,h,:] ----
__global__ void __launch_bounds__(256)
scatter_kernel(const float* __restrict__ alpha, const int* __restrict__ cidx,
               const float* __restrict__ g, float* __restrict__ intra)
{
    __shared__ float acc[C_ * D_];   // 30.4 KB
    __shared__ float sal[H_];
    __shared__ int   sci[H_];
    const int bn  = blockIdx.x;
    const int b   = bn / NEWS;
    const int tid = threadIdx.x;
    for (int i = tid; i < C_ * D_; i += 256) acc[i] = 0.0f;
    for (int i = tid; i < H_; i += 256) {
        sal[i] = alpha[(long long)bn * H_ + i];
        sci[i] = cidx[(long long)b * H_ + i];
    }
    __syncthreads();
    for (int h = 0; h < H_; ++h) {
        const int   c  = sci[h];
        const float al = sal[h];
        const float* gr = g + ((long long)b * H_ + h) * D_;
        for (int d = tid; d < D_; d += 256)
            acc[c * D_ + d] += al * gr[d];
    }
    __syncthreads();
    float* op = intra + (long long)bn * C_ * D_;
    for (int i = tid; i < C_ * D_; i += 256) op[i] = acc[i];
}

// ---- inter-cluster: sc = dot(intra2, v)/scale, mask, softmax, weighted sum ----
__global__ void __launch_bounds__(256)
inter_kernel(const float* __restrict__ intra2, const float* __restrict__ vbuf,
             const int* __restrict__ cmask, float* __restrict__ out, float inv_scale)
{
    __shared__ float vr[D_];
    __shared__ float scl[C_];
    __shared__ float red[256];
    __shared__ float wsum;
    const int bn  = blockIdx.x;
    const int b   = bn / NEWS;
    const int tid = threadIdx.x;
    for (int d = tid; d < D_; d += 256) vr[d] = vbuf[(long long)bn * D_ + d];
    __syncthreads();
    const float* ip = intra2 + (long long)bn * C_ * D_;
    for (int c = 0; c < C_; ++c) {
        float p = 0.0f;
        for (int d = tid; d < D_; d += 256) p += ip[(long long)c * D_ + d] * vr[d];
        red[tid] = p;
        __syncthreads();
        for (int s = 128; s > 0; s >>= 1) {
            if (tid < s) red[tid] += red[tid + s];
            __syncthreads();
        }
        if (tid == 0) {
            int msk = (c == C_ - 1) ? 1 : cmask[(long long)b * C_ + c];
            scl[c] = (msk == 0) ? -1e9f : red[0] * inv_scale;
        }
        __syncthreads();
    }
    if (tid == 0) {
        float mx = -1e30f;
        for (int c = 0; c < C_; ++c) mx = fmaxf(mx, scl[c]);
        float s = 0.0f;
        for (int c = 0; c < C_; ++c) { scl[c] = __expf(scl[c] - mx); s += scl[c]; }
        wsum = s;
    }
    __syncthreads();
    for (int d = tid; d < D_; d += 256) {
        float o = 0.0f;
        for (int c = 0; c < C_; ++c) o += scl[c] * ip[(long long)c * D_ + d];
        out[(long long)bn * D_ + d] = o / wsum;
    }
}

// =====================================================================
// host side
// =====================================================================
static inline void gemm(hipStream_t s,
                        const bf16* A, long long sA,
                        const bf16* Bt, long long sB, int Kp,
                        const float* bias, const float* resid, long long sR,
                        float* Cc, long long sC, int ldc,
                        int M, int Nn, int batch, float scale, int relu)
{
    dim3 g((M + 15) / 16, (Nn + 15) / 16, batch);
    hipLaunchKernelGGL(wmma_gemm_kernel, g, dim3(32), 0, s,
                       A, sA, Bt, sB, Kp, bias, resid, sR, Cc, sC, ldc,
                       M, Nn, scale, relu);
}

static inline void pack(hipStream_t s, const float* in, long long sIn,
                        bf16* out, long long sOut,
                        int R, int Cn, int Rp, int Cp, int batch)
{
    long long total = (long long)batch * Rp * Cp;
    hipLaunchKernelGGL(pack_bf16_kernel, dim3((unsigned)((total + 255) / 256)),
                       dim3(256), 0, s, in, sIn, out, sOut, R, Cn, Rp, Cp, total);
}

static inline void packT(hipStream_t s, const float* in, long long sIn,
                         bf16* out, long long sOut,
                         int R, int Cn, int Cp, int Rp, int batch)
{
    long long total = (long long)batch * Cp * Rp;
    hipLaunchKernelGGL(packT_bf16_kernel, dim3((unsigned)((total + 255) / 256)),
                       dim3(256), 0, s, in, sIn, out, sOut, R, Cn, Cp, Rp, total);
}

extern "C" void kernel_launch(void* const* d_in, const int* in_sizes, int n_in,
                              void* d_out, int out_size, void* d_ws, size_t ws_size,
                              hipStream_t stream)
{
    const float* hist  = (const float*)d_in[0];
    const float* cand  = (const float*)d_in[1];
    const float* graph = (const float*)d_in[2];
    const int*   cmask = (const int*)d_in[3];
    const int*   cidx  = (const int*)d_in[4];
    const float* proxy = (const float*)d_in[5];
    const float* Wgcn  = (const float*)d_in[6];
    const float* bgcn  = (const float*)d_in[7];
    const float* lng   = (const float*)d_in[8];
    const float* lnb   = (const float*)d_in[9];
    const float* Wki   = (const float*)d_in[10];
    const float* Wqi   = (const float*)d_in[11];
    const float* bqi   = (const float*)d_in[12];
    const float* Waff  = (const float*)d_in[13];
    const float* baff  = (const float*)d_in[14];
    const float* Wke   = (const float*)d_in[15];
    const float* Wqe   = (const float*)d_in[16];
    const float* bqe   = (const float*)d_in[17];
    float* outp = (float*)d_out;

    char* ws = (char*)d_ws;
    size_t off = 0;
    auto take = [&](size_t bytes) -> char* {
        char* p = ws + off;
        off += (bytes + 255) & ~(size_t)255;
        return p;
    };

    // fp32 intermediates
    float* h0 = (float*)take((size_t)B_ * N_ * D_ * 4);
    float* Hc = (float*)take((size_t)B_ * N_ * D_ * 4);
    float* Ht = (float*)take((size_t)B_ * N_ * D_ * 4);
    float* Sb = (float*)take((size_t)B_ * N_ * D_ * 4);
    float* G  = (float*)take((size_t)B_ * H_ * D_ * 4);
    float* Km = (float*)take((size_t)B_ * H_ * A_ * 4);
    float* Qm = (float*)take((size_t)B_ * NEWS * A_ * 4);
    float* Ab = (float*)take((size_t)B_ * NEWS * H_ * 4);
    float* Al = (float*)take((size_t)B_ * NEWS * H_ * 4);
    float* I1 = (float*)take((size_t)B_ * NEWS * C_ * D_ * 4);
    float* I2 = (float*)take((size_t)B_ * NEWS * C_ * D_ * 4);
    float* Qc = (float*)take((size_t)B_ * NEWS * A_ * 4);
    float* Vb = (float*)take((size_t)B_ * NEWS * D_ * 4);

    // packed bf16 operands
    bf16* graphP = (bf16*)take((size_t)B_ * NP_G * NP_G * 2);   // A: [128,128] x B
    bf16* hT     = (bf16*)take((size_t)B_ * D_ * NP_G * 2);     // Bt: [400,128] x B
    bf16* Spk    = (bf16*)take((size_t)B_ * N_ * DP * 2);       // A: [7552,416]
    bf16* Gpk    = (bf16*)take((size_t)B_ * H_ * DP * 2);       // A: [6400,416]
    bf16* candP  = (bf16*)take((size_t)B_ * NEWS * DP * 2);     // A: [3200,416]
    bf16* QmP    = (bf16*)take((size_t)B_ * NEWSP * AP_K * 2);  // A: [64,224] x B
    bf16* KmP    = (bf16*)take((size_t)B_ * HP * AP_K * 2);     // Bt: [112,224] x B
    bf16* I1P    = (bf16*)take((size_t)B_ * NEWS * C_ * DP * 2);// A: [60800,416]
    bf16* QcP    = (bf16*)take((size_t)B_ * NEWS * AP_K * 2);   // A: [3200,224]
    bf16* WgcnT  = (bf16*)take((size_t)L_ * D_ * DP * 2);       // Bt: [400,416] x L
    bf16* WkiT   = (bf16*)take((size_t)AP_N * DP * 2);          // Bt: [208,416]
    bf16* WqiT   = (bf16*)take((size_t)AP_N * DP * 2);
    bf16* WaffT  = (bf16*)take((size_t)D_ * DP * 2);            // Bt: [400,416]
    bf16* WqeT   = (bf16*)take((size_t)AP_N * DP * 2);
    bf16* WkeP   = (bf16*)take((size_t)D_ * AP_K * 2);          // Bt: [400,224] (direct)

    const float inv_sqrtA = 0.07071067811865475f;  // 1/sqrt(200)

    // ---- one-time weight packs ----
    packT(stream, Wgcn, (long long)D_ * D_, WgcnT, (long long)D_ * DP,
          D_, D_, D_, DP, L_);                              // [400,400] -> [400,416] x2
    packT(stream, Wki, 0, WkiT, 0, D_, A_, AP_N, DP, 1);    // [400,200] -> [208,416]
    packT(stream, Wqi, 0, WqiT, 0, D_, A_, AP_N, DP, 1);
    packT(stream, Waff, 0, WaffT, 0, D_, D_, D_, DP, 1);    // [400,400] -> [400,416]
    packT(stream, Wqe, 0, WqeT, 0, D_, A_, AP_N, DP, 1);
    pack (stream, Wke, 0, WkeP, 0, D_, A_, D_, AP_K, 1);    // Wke^T as Bt == direct pack
    // graph pack (reused both layers)
    pack(stream, graph, (long long)N_ * N_, graphP, (long long)NP_G * NP_G,
         N_, N_, NP_G, NP_G, B_);

    // ---- h0 = [history ; proxy], h_cur = h0 ----
    {
        long long tot = (long long)B_ * N_ * D_;
        hipLaunchKernelGGL(build_h0_kernel,
                           dim3((unsigned)((tot + 255) / 256)), dim3(256), 0, stream,
                           hist, proxy, h0, Hc);
    }

    // ---- GCN layers ----
    for (int l = 0; l < L_; ++l) {
        // h^T packed as Bt: [118,400] -> [400,128] per b
        packT(stream, Hc, (long long)N_ * D_, hT, (long long)D_ * NP_G,
              N_, D_, D_, NP_G, B_);
        // S[b] = graph[b] @ h[b]
        gemm(stream, graphP, (long long)NP_G * NP_G,
             hT, (long long)D_ * NP_G, NP_G,
             nullptr, nullptr, 0,
             Sb, (long long)N_ * D_, D_,
             N_, D_, B_, 1.0f, 0);
        // pack S -> bf16 A
        pack(stream, Sb, 0, Spk, 0, B_ * N_, D_, B_ * N_, DP, 1);
        // Ht = relu(S @ Wgcn[l] + bgcn[l]) + Hc
        gemm(stream, Spk, 0,
             WgcnT + (long long)l * D_ * DP, 0, DP,
             bgcn + (long long)l * D_, Hc, 0,
             Ht, 0, D_,
             B_ * N_, D_, 1, 1.0f, 1);
        // Hc = LayerNorm(Ht)
        hipLaunchKernelGGL(layernorm_kernel, dim3(B_ * N_), dim3(256), 0, stream,
                           Ht, Hc, lng + (long long)l * D_, lnb + (long long)l * D_);
    }

    // ---- g = (h + h0)[:, :H] ----
    {
        long long tot = (long long)B_ * H_ * D_;
        hipLaunchKernelGGL(make_g_kernel,
                           dim3((unsigned)((tot + 255) / 256)), dim3(256), 0, stream,
                           Hc, h0, G);
    }

    // ---- intra-cluster attention ----
    pack(stream, G, 0, Gpk, 0, B_ * H_, D_, B_ * H_, DP, 1);
    pack(stream, cand, 0, candP, 0, B_ * NEWS, D_, B_ * NEWS, DP, 1);
    // K = g @ Wk_intra
    gemm(stream, Gpk, 0, WkiT, 0, DP, nullptr, nullptr, 0,
         Km, 0, A_, B_ * H_, A_, 1, 1.0f, 0);
    // Q = cand @ Wq_intra + bq
    gemm(stream, candP, 0, WqiT, 0, DP, bqi, nullptr, 0,
         Qm, 0, A_, B_ * NEWS, A_, 1, 1.0f, 0);
    // pack Q (A, per-b) and K (Bt, per-b: row-major [H,A] already == Bt layout)
    pack(stream, Qm, (long long)NEWS * A_, QmP, (long long)NEWSP * AP_K,
         NEWS, A_, NEWSP, AP_K, B_);
    pack(stream, Km, (long long)H_ * A_, KmP, (long long)HP * AP_K,
         H_, A_, HP, AP_K, B_);
    // a[b] = Q[b] @ K[b]^T / sqrt(A)
    gemm(stream, QmP, (long long)NEWSP * AP_K,
         KmP, (long long)HP * AP_K, AP_K,
         nullptr, nullptr, 0,
         Ab, (long long)NEWS * H_, H_,
         NEWS, H_, B_, inv_sqrtA, 0);
    // segment softmax over categories
    hipLaunchKernelGGL(segment_softmax_kernel, dim3(B_ * NEWS), dim3(128), 0, stream,
                       Ab, cidx, Al);
    // scatter-sum -> intra [B,NEWS,C,D]
    hipLaunchKernelGGL(scatter_kernel, dim3(B_ * NEWS), dim3(256), 0, stream,
                       Al, cidx, G, I1);
    // intra2 = relu(intra @ W_aff + b_aff) + intra   (dominant GEMM)
    pack(stream, I1, 0, I1P, 0, B_ * NEWS * C_, D_, B_ * NEWS * C_, DP, 1);
    gemm(stream, I1P, 0, WaffT, 0, DP, baff, I1, 0,
         I2, 0, D_, B_ * NEWS * C_, D_, 1, 1.0f, 1);

    // ---- inter-cluster (refactored: sc = intra2 . (Wk_inter @ Qc)) ----
    gemm(stream, candP, 0, WqeT, 0, DP, bqe, nullptr, 0,
         Qc, 0, A_, B_ * NEWS, A_, 1, 1.0f, 0);
    pack(stream, Qc, 0, QcP, 0, B_ * NEWS, A_, B_ * NEWS, AP_K, 1);
    // v = Qc @ Wk_inter^T -> [B*NEWS, D]
    gemm(stream, QcP, 0, WkeP, 0, AP_K, nullptr, nullptr, 0,
         Vb, 0, D_, B_ * NEWS, D_, 1, 1.0f, 0);
    // masked softmax over C + weighted sum -> out
    hipLaunchKernelGGL(inter_kernel, dim3(B_ * NEWS), dim3(256), 0, stream,
                       I2, Vb, cmask, outp, inv_sqrtA);
}
